// Network_68650757259534
// MI455X (gfx1250) — matrix operand order repaired and checked by reading, not compiled, not run on hardware
//
#include <hip/hip_runtime.h>
#include <hip/hip_bf16.h>

typedef _Float16 h16;
typedef __attribute__((ext_vector_type(16))) _Float16 v16h;
typedef __attribute__((ext_vector_type(8)))  float    v8f;

union Frag16 { v16h v; unsigned int u[8]; h16 e[16]; };
union Pack8h { h16 h[8]; uint4 q; };

// A fragment: 16(M) x 32(K) f16, W row-major (O, lda). contiguous k order (same as B):
// lanes 0-15 (M=0..15) hold K=0..15 (VGPR0-7); lanes 16-31: K=16..31.
__device__ __forceinline__ v16h load_a_frag(const h16* __restrict__ W, int lda,
                                            int mbase, int k0, int oclamp) {
  Frag16 f;
  int lane = threadIdx.x & 31;
  int m = lane & 15, kh = lane >> 4;
  int row = mbase + m; row = row > oclamp ? oclamp : row;
  const unsigned int* p = (const unsigned int*)(W + (size_t)row * lda) + (k0 >> 1) + kh * 8;
#pragma unroll
  for (int i = 0; i < 4; ++i) { f.u[i] = p[i]; f.u[i + 4] = p[i + 4]; }
  return f.v;
}

// B fragment from TRANSPOSED activation layout XT[n][k] (row stride ldk): 2x b128 per lane.
__device__ __forceinline__ v16h load_b_fragT(const h16* __restrict__ XT, int ldk,
                                             int k0, int nbase) {
  Frag16 f;
  int lane = threadIdx.x & 31;
  int n = nbase + (lane & 15), kh = lane >> 4;
  const unsigned int* p = (const unsigned int*)(XT + (size_t)n * ldk + k0) + kh * 8;
#pragma unroll
  for (int i = 0; i < 8; ++i) f.u[i] = p[i];
  return f.v;
}

// Shared epilogue. ymode 1: transposed store (n-major, ldy) with 16B vector stores;
// ymode 0: channel-major scalar stores. colsum: acc *= 1/(1e-9+colsum[n]).
// residMode: 0 none | 1 Y=resid+act(bn(acc+bias)) | 2 Y=resid-acc.
__device__ __forceinline__ void epilogue(const v8f* cc, int bi, int n0, int mt,
    h16* __restrict__ Yh, float* __restrict__ Yf, long ystride, int ldy, int ymode,
    const float* __restrict__ bnG, const float* __restrict__ bnB,
    const float* __restrict__ bnM, const float* __restrict__ bnV,
    const float* __restrict__ biasO, const float* __restrict__ biasBO,
    const float* __restrict__ colsum,
    const h16* __restrict__ resid, long rstride, int ldr, int residMode,
    int act, int O, int NN) {
  int lane = threadIdx.x & 31;
  int nl = lane & 15, mh = lane >> 4;
  int o0 = (mt << 4) + (mh << 3);   // C/D layout: VGPR r -> M = r + 8*(lane>=16)
#pragma unroll
  for (int j = 0; j < 4; ++j) {
    int n = n0 + j * 16 + nl;
    float csv = colsum ? 1.0f / (1e-9f + colsum[(size_t)bi * NN + n]) : 1.0f;
    Pack8h rsd;
    if (resid) rsd.q = *(const uint4*)(resid + (size_t)bi * rstride + (size_t)n * ldr + o0);
    float val[8];
#pragma unroll
    for (int r = 0; r < 8; ++r) {
      int o = o0 + r;
      float v = cc[j][r] * csv;
      if (residMode == 2) {
        v = (float)rsd.h[r] - v;
      } else {
        bool ok = o < O;
        if (biasO && ok)  v += biasO[o];
        if (biasBO && ok) v += biasBO[bi * O + o];
        if (bnG && ok) {
          float sc = bnG[o] * rsqrtf(bnV[o] + 1e-5f);
          v = (v - bnM[o]) * sc + bnB[o];
        }
        if (act == 1)      v = v > 0.f ? v : 0.f;
        else if (act == 2) v = v > 0.f ? v : 0.2f * v;
        if (residMode == 1) v += (float)rsd.h[r];
      }
      val[r] = v;
    }
    if (ymode) {
      size_t base = (size_t)bi * ystride + (size_t)n * ldy + o0;
      if (Yh) {
        if (o0 + 8 <= O) {
          Pack8h st;
#pragma unroll
          for (int r = 0; r < 8; ++r) st.h[r] = (h16)val[r];
          *(uint4*)(Yh + base) = st.q;
        } else {
#pragma unroll
          for (int r = 0; r < 8; ++r) if (o0 + r < O) Yh[base + r] = (h16)val[r];
        }
      }
      if (Yf) {
#pragma unroll
        for (int r = 0; r < 8; ++r) if (o0 + r < O) Yf[base + r] = val[r];
      }
    } else {
#pragma unroll
      for (int r = 0; r < 8; ++r) {
        int o = o0 + r;
        if (o >= O) continue;
        size_t iy = (size_t)bi * ystride + (size_t)o * NN + n;
        if (Yh) Yh[iy] = (h16)val[r];
        if (Yf) Yf[iy] = val[r];
      }
    }
  }
}

// ---------------- per-wave GEMM (small O: qk O=32, head O=50) ----------------
__global__ __launch_bounds__(256)
void gemm_wmma_kernel(const h16* __restrict__ A, long astride,
                      const h16* __restrict__ X, long xstride, int ldk,
                      h16* __restrict__ Yh, float* __restrict__ Yf,
                      long ystride, int ldy, int ymode,
                      const float* __restrict__ bnG, const float* __restrict__ bnB,
                      const float* __restrict__ bnM, const float* __restrict__ bnV,
                      const float* __restrict__ biasO, const float* __restrict__ biasBO,
                      const float* __restrict__ colsum,
                      const h16* __restrict__ resid, long rstride, int ldr, int residMode,
                      int act, int O, int I, int NN, int nbatch) {
  int wave = blockIdx.x * 8 + (threadIdx.x >> 5);
  int mtiles = (O + 15) >> 4;
  int ntiles = NN >> 6;
  int total = nbatch * mtiles * ntiles;
  if (wave >= total) return;
  int bi  = wave / (mtiles * ntiles);
  int rem = wave % (mtiles * ntiles);
  int mt = rem / ntiles, nt = rem % ntiles;

  const h16* Ab = A + (size_t)bi * astride;
  const h16* Xb = X + (size_t)bi * xstride;
  int n0 = nt << 6;

  v8f c0 = {}, c1 = {}, c2 = {}, c3 = {};
  for (int k0 = 0; k0 < I; k0 += 32) {
    if (k0 + 32 < I)
      __builtin_prefetch((const void*)(Xb + (size_t)n0 * ldk + k0 + 32), 0, 3);
    v16h af = load_a_frag(Ab, I, mt << 4, k0, O - 1);
    v16h b0 = load_b_fragT(Xb, ldk, k0, n0);
    v16h b1 = load_b_fragT(Xb, ldk, k0, n0 + 16);
    v16h b2 = load_b_fragT(Xb, ldk, k0, n0 + 32);
    v16h b3 = load_b_fragT(Xb, ldk, k0, n0 + 48);
    c0 = __builtin_amdgcn_wmma_f32_16x16x32_f16(false, af, false, b0, (short)0, c0, false, false);
    c1 = __builtin_amdgcn_wmma_f32_16x16x32_f16(false, af, false, b1, (short)0, c1, false, false);
    c2 = __builtin_amdgcn_wmma_f32_16x16x32_f16(false, af, false, b2, (short)0, c2, false, false);
    c3 = __builtin_amdgcn_wmma_f32_16x16x32_f16(false, af, false, b3, (short)0, c3, false, false);
  }
  v8f cc[4] = {c0, c1, c2, c3};
  epilogue(cc, bi, n0, mt, Yh, Yf, ystride, ldy, ymode, bnG, bnB, bnM, bnV,
           biasO, biasBO, colsum, resid, rstride, ldr, residMode, act, O, NN);
}

// ------------- workgroup-cooperative GEMM with async B staging in LDS --------------
// Requires O % 128 == 0. One WG (8 waves) = 128(M) x 64(N) tile; the 64x32 f16 B panel
// is staged once per K-step via GLOBAL_LOAD_ASYNC_TO_LDS_B128 (double buffered, ASYNCcnt),
// then consumed by all 8 waves as ds_load_b128 fragments. 8x less global B traffic.
#define LDS_ROW 40  // halves per row (32 data + 8 pad -> 80B stride, bank-conflict stagger)
__global__ __launch_bounds__(256)
void gemm_wmma_lds_kernel(const h16* __restrict__ A, long astride,
                          const h16* __restrict__ X, long xstride, int ldk,
                          h16* __restrict__ Yh, float* __restrict__ Yf,
                          long ystride, int ldy, int ymode,
                          const float* __restrict__ bnG, const float* __restrict__ bnB,
                          const float* __restrict__ bnM, const float* __restrict__ bnV,
                          const float* __restrict__ biasO, const float* __restrict__ biasBO,
                          const float* __restrict__ colsum,
                          const h16* __restrict__ resid, long rstride, int ldr, int residMode,
                          int act, int O, int I, int NN, int nbatch) {
  __shared__ __align__(16) h16 smem[2][64 * LDS_ROW];

  int mgroups = O >> 7;           // mtiles / 8
  int ntiles = NN >> 6;
  int wg = blockIdx.x;
  int bi  = wg / (mgroups * ntiles);
  int rem = wg % (mgroups * ntiles);
  int mg = rem / ntiles, nt = rem % ntiles;
  int wv = threadIdx.x >> 5;
  int mt = mg * 8 + wv;           // this wave's 16-row M tile
  int n0 = nt << 6;

  const h16* Ab = A + (size_t)bi * astride;
  const h16* Xb = X + (size_t)bi * xstride;

  // staging assignment: thread t copies 16B: row nr = t/4 (n0+nr), chunk ck = t%4 (halves ck*8..+7)
  int t = threadIdx.x;
  int nr = t >> 2, ck = t & 3;
  const h16* gsrc = Xb + (size_t)(n0 + nr) * ldk + ck * 8;
  unsigned ldsOff0 = (unsigned)(uintptr_t)&smem[0][nr * LDS_ROW + ck * 8];
  unsigned ldsOff1 = (unsigned)(uintptr_t)&smem[1][nr * LDS_ROW + ck * 8];

  auto stage = [&](int s, int k0) {
    unsigned long long ga = (unsigned long long)(uintptr_t)(gsrc + k0);
    unsigned la = s ? ldsOff1 : ldsOff0;
    asm volatile("global_load_async_to_lds_b128 %0, %1, off"
                 :: "v"(la), "v"(ga) : "memory");
  };

  stage(0, 0);
  asm volatile("s_wait_asynccnt 0x0" ::: "memory");
  __syncthreads();

  int lane = threadIdx.x & 31;
  int nl = lane & 15, kh = lane >> 4;

  v8f c0 = {}, c1 = {}, c2 = {}, c3 = {};
  for (int k0 = 0; k0 < I; k0 += 32) {
    int s = (k0 >> 5) & 1;
    if (k0 + 32 < I) stage(s ^ 1, k0 + 32);   // overlap next panel copy with WMMAs

    v16h af = load_a_frag(Ab, I, mt << 4, k0, O - 1);
    const h16* base = &smem[s][0];
    Frag16 f0, f1, f2, f3;
    {
      const h16* rp = base + (size_t)(0 * 16 + nl) * LDS_ROW + kh * 16;
      *(uint4*)&f0.u[0] = *(const uint4*)rp; *(uint4*)&f0.u[4] = *(const uint4*)(rp + 8);
      rp = base + (size_t)(1 * 16 + nl) * LDS_ROW + kh * 16;
      *(uint4*)&f1.u[0] = *(const uint4*)rp; *(uint4*)&f1.u[4] = *(const uint4*)(rp + 8);
      rp = base + (size_t)(2 * 16 + nl) * LDS_ROW + kh * 16;
      *(uint4*)&f2.u[0] = *(const uint4*)rp; *(uint4*)&f2.u[4] = *(const uint4*)(rp + 8);
      rp = base + (size_t)(3 * 16 + nl) * LDS_ROW + kh * 16;
      *(uint4*)&f3.u[0] = *(const uint4*)rp; *(uint4*)&f3.u[4] = *(const uint4*)(rp + 8);
    }
    c0 = __builtin_amdgcn_wmma_f32_16x16x32_f16(false, af, false, f0.v, (short)0, c0, false, false);
    c1 = __builtin_amdgcn_wmma_f32_16x16x32_f16(false, af, false, f1.v, (short)0, c1, false, false);
    c2 = __builtin_amdgcn_wmma_f32_16x16x32_f16(false, af, false, f2.v, (short)0, c2, false, false);
    c3 = __builtin_amdgcn_wmma_f32_16x16x32_f16(false, af, false, f3.v, (short)0, c3, false, false);

    asm volatile("s_wait_asynccnt 0x0" ::: "memory");  // own copies done
    __syncthreads();                                   // whole WG's panel visible
  }
  v8f cc[4] = {c0, c1, c2, c3};
  epilogue(cc, bi, n0, mt, Yh, Yf, ystride, ldy, ymode, bnG, bnB, bnM, bnV,
           biasO, biasBO, colsum, resid, rstride, ldr, residMode, act, O, NN);
}

__global__ void cvt_f16_kernel(const float* __restrict__ src, h16* __restrict__ dst,
                               int rows, int cols, int srcld) {
  int idx = blockIdx.x * blockDim.x + threadIdx.x;
  if (idx >= rows * cols) return;
  int r = idx / cols, c = idx % cols;
  dst[idx] = (h16)src[(size_t)r * srcld + c];
}

__global__ void stem_kernel(const float* __restrict__ pts, const float* __restrict__ w1,
                            const float* g, const float* be, const float* mu, const float* va,
                            h16* __restrict__ out, int nbatch, int NN) {
  int idx = blockIdx.x * blockDim.x + threadIdx.x;
  if (idx >= nbatch * 128 * NN) return;
  int c = idx & 127; int n = (idx >> 7) % NN; int b = idx / (NN * 128);
  const float* pb = pts + (size_t)b * 3 * NN + n;
  float s = w1[c * 3 + 0] * pb[0] + w1[c * 3 + 1] * pb[NN] + w1[c * 3 + 2] * pb[2 * NN];
  float sc = g[c] * rsqrtf(va[c] + 1e-5f);
  s = (s - mu[c]) * sc + be[c];
  out[idx] = (h16)(s > 0.f ? s : 0.f);
}

__global__ __launch_bounds__(256)
void rowstats_kernel(const h16* __restrict__ E, float* __restrict__ rmax,
                     float* __restrict__ rinv, int NN, int nbatch) {
  int row = blockIdx.x * 8 + (threadIdx.x >> 5);
  if (row >= nbatch * NN) return;
  int lane = threadIdx.x & 31;
  const h16* p = E + (size_t)row * NN;
  float mx = -1e30f;
  for (int m = lane; m < NN; m += 32) mx = fmaxf(mx, (float)p[m]);
  for (int off = 16; off; off >>= 1) mx = fmaxf(mx, __shfl_xor(mx, off, 32));
  float s = 0.f;
  for (int m = lane; m < NN; m += 32) s += __expf((float)p[m] - mx);
  for (int off = 16; off; off >>= 1) s += __shfl_xor(s, off, 32);
  rmax[row] = mx;
  rinv[row] = 1.0f / s;
}

// Symmetric E: overwrite row m with attnT[m][n]=exp(E[m][n]-rmax[n])*rinv[n]; csb[m]=row sum.
__global__ __launch_bounds__(256)
void softmax_applyT_kernel(h16* __restrict__ E, const float* __restrict__ rmax,
                           const float* __restrict__ rinv, float* __restrict__ csb,
                           int NN, int nbatch) {
  int row = blockIdx.x * 8 + (threadIdx.x >> 5);
  if (row >= nbatch * NN) return;
  int lane = threadIdx.x & 31;
  int b = row / NN;
  h16* p = E + (size_t)row * NN;
  const float* rm = rmax + (size_t)b * NN;
  const float* ri = rinv + (size_t)b * NN;
  float s = 0.f;
  for (int n = lane; n < NN; n += 32) {
    float a = __expf((float)p[n] - rm[n]) * ri[n];
    p[n] = (h16)a;
    s += a;
  }
  for (int off = 16; off; off >>= 1) s += __shfl_xor(s, off, 32);
  csb[row] = s;
}

__global__ void pool_kernel(const h16* __restrict__ xfT, float* __restrict__ xmax,
                            float* __restrict__ xavg, int Cc, int NN, int nbatch) {
  int idx = blockIdx.x * blockDim.x + threadIdx.x;
  if (idx >= nbatch * Cc) return;
  int b = idx / Cc, c = idx % Cc;
  const h16* p = xfT + (size_t)b * NN * Cc + c;
  float mx = -1e30f, s = 0.f;
  for (int n = 0; n < NN; ++n) { float v = (float)p[(size_t)n * Cc]; mx = fmaxf(mx, v); s += v; }
  xmax[idx] = mx; xavg[idx] = s / (float)NN;
}

__global__ void label_kernel(const float* __restrict__ cls, const float* __restrict__ wl,
                             const float* g, const float* be, const float* mu, const float* va,
                             float* __restrict__ lbl, int nbatch) {
  int idx = blockIdx.x * blockDim.x + threadIdx.x;
  if (idx >= nbatch * 64) return;
  int c = idx & 63, b = idx >> 6;
  float s = 0.f;
  for (int j = 0; j < 16; ++j) s += wl[c * 16 + j] * cls[b * 16 + j];
  float sc = g[c] * rsqrtf(va[c] + 1e-5f);
  s = (s - mu[c]) * sc + be[c];
  lbl[idx] = s > 0.f ? s : 0.2f * s;
}

__global__ void bias2_kernel(const float* __restrict__ ws1, const float* __restrict__ bs1,
                             const float* __restrict__ xmax, const float* __restrict__ xavg,
                             const float* __restrict__ lbl, float* __restrict__ bias2,
                             int nbatch) {
  int idx = blockIdx.x * blockDim.x + threadIdx.x;
  if (idx >= nbatch * 512) return;
  int o = idx & 511, b = idx >> 9;
  const float* wr = ws1 + (size_t)o * 3136;
  float s = bs1[o];
  for (int c = 0; c < 1024; ++c)
    s += wr[1024 + c] * xmax[b * 1024 + c] + wr[2048 + c] * xavg[b * 1024 + c];
  for (int j = 0; j < 64; ++j) s += wr[3072 + j] * lbl[b * 64 + j];
  bias2[idx] = s;
}

__global__ void logsoftmax_kernel(const float* __restrict__ y3, float* __restrict__ out,
                                  int NN, int nbatch) {
  int idx = blockIdx.x * blockDim.x + threadIdx.x;
  if (idx >= nbatch * NN) return;
  const float* p = y3 + (size_t)idx * 50;
  float mx = -1e30f;
  for (int o = 0; o < 50; ++o) mx = fmaxf(mx, p[o]);
  float s = 0.f;
  for (int o = 0; o < 50; ++o) s += __expf(p[o] - mx);
  float lse = mx + __logf(s);
  float* q = out + (size_t)idx * 50;
  for (int o = 0; o < 50; ++o) q[o] = p[o] - lse;
}

extern "C" void kernel_launch(void* const* d_in, const int* in_sizes, int n_in,
                              void* d_out, int out_size, void* d_ws, size_t ws_size,
                              hipStream_t stream) {
  (void)in_sizes; (void)n_in; (void)out_size; (void)ws_size;
  constexpr int B = 8, N = 2048, C = 128;
  auto F = [&](int k) { return (const float*)d_in[k]; };

  const float* points = F(0);
  const float* cls    = F(1);
  const float* w1 = F(2);
  const float *bn1g = F(3), *bn1b = F(4), *bn1m = F(5), *bn1v = F(6);
  const float* w2 = F(7);
  const float *bn2g = F(8), *bn2b = F(9), *bn2m = F(10), *bn2v = F(11);
  const float *wqk[4], *wv[4], *wt[4], *bg[4], *bbt[4], *bm[4], *bv[4];
  for (int i = 0; i < 4; ++i) {
    int base = 12 + 7 * i;
    wqk[i] = F(base); wv[i] = F(base + 1); wt[i] = F(base + 2);
    bg[i] = F(base + 3); bbt[i] = F(base + 4); bm[i] = F(base + 5); bv[i] = F(base + 6);
  }
  const float* wf = F(40);
  const float *bnfg = F(41), *bnfb = F(42), *bnfm = F(43), *bnfv = F(44);
  const float* wl = F(45);
  const float *bnlg = F(46), *bnlb = F(47), *bnlm = F(48), *bnlv = F(49);
  const float* ws1 = F(50); const float* bs1 = F(51);
  const float *bns1g = F(52), *bns1b = F(53), *bns1m = F(54), *bns1v = F(55);
  const float* ws2 = F(56); const float* bs2 = F(57);
  const float *bns2g = F(58), *bns2b = F(59), *bns2m = F(60), *bns2v = F(61);
  const float* ws3 = F(62); const float* bs3 = F(63);

  // ---- workspace ----
  char* wsb = (char*)d_ws;
  size_t off = 0;
  auto alloc = [&](size_t bytes) -> void* {
    off = (off + 255) & ~(size_t)255;
    void* p = wsb + off; off += bytes; return p;
  };
  const size_t NB = (size_t)B;
  h16* x1T   = (h16*)alloc(NB * N * C * 2);     // (b, n, 128)
  h16* x0T   = (h16*)alloc(NB * N * C * 2);
  h16* xcatT = (h16*)alloc(NB * N * 512 * 2);   // (b, n, 512)
  h16* qkT   = (h16*)alloc(NB * N * 32 * 2);    // (b, n, 32)
  h16* vh    = (h16*)alloc(NB * C * N * 2);     // (b, 128, n) channel-major
  h16* tT    = (h16*)alloc(NB * N * C * 2);
  float* rmax = (float*)alloc(NB * N * 4);
  float* rinv = (float*)alloc(NB * N * 4);
  float* csb  = (float*)alloc(NB * N * 4);
  float* xmax = (float*)alloc(NB * 1024 * 4);
  float* xavg = (float*)alloc(NB * 1024 * 4);
  float* lblf = (float*)alloc(NB * 64 * 4);
  float* bias2 = (float*)alloc(NB * 512 * 4);
  h16* w2h = (h16*)alloc(128 * 128 * 2);
  h16 *wqkh[4], *wvh[4], *wth[4];
  for (int i = 0; i < 4; ++i) {
    wqkh[i] = (h16*)alloc(32 * 128 * 2);
    wvh[i]  = (h16*)alloc(128 * 128 * 2);
    wth[i]  = (h16*)alloc(128 * 128 * 2);
  }
  h16* wfh   = (h16*)alloc(1024 * 512 * 2);
  h16* ws1ah = (h16*)alloc(512 * 1024 * 2);
  h16* ws2h  = (h16*)alloc(256 * 512 * 2);
  h16* ws3h  = (h16*)alloc(50 * 256 * 2);
  h16* Eb    = (h16*)alloc(NB * N * N * 2);     // 64 MB; reused post-attention:
  char* post = (char*)Eb;
  h16* xfT = (h16*)post;  post += NB * N * 1024 * 2;
  h16* y1T = (h16*)post;  post += NB * N * 512 * 2;
  h16* y2T = (h16*)post;  post += NB * N * 256 * 2;
  float* y3T = (float*)post;                    // (b, n, 50) f32

  auto cvt = [&](const float* src, h16* dst, int rows, int cols, int srcld) {
    int cnt = rows * cols;
    cvt_f16_kernel<<<(cnt + 255) / 256, 256, 0, stream>>>(src, dst, rows, cols, srcld);
  };
  auto gemm = [&](const h16* A, long astride, const h16* X, long xstride, int ldk,
                  h16* Yh, float* Yf, long ystride, int ldy, int ymode,
                  const float* g, const float* be, const float* mu, const float* va,
                  const float* biasO, const float* biasBO, const float* colsum,
                  const h16* resid, long rstride, int ldr, int residMode, int act,
                  int O, int I, int NN, int nb) {
    if ((O & 127) == 0) {
      int blocks = nb * (O >> 7) * (NN >> 6);
      gemm_wmma_lds_kernel<<<blocks, 256, 0, stream>>>(
          A, astride, X, xstride, ldk, Yh, Yf, ystride, ldy, ymode, g, be, mu, va,
          biasO, biasBO, colsum, resid, rstride, ldr, residMode, act, O, I, NN, nb);
    } else {
      int waves = nb * ((O + 15) >> 4) * (NN >> 6);
      gemm_wmma_kernel<<<(waves + 7) >> 3, 256, 0, stream>>>(
          A, astride, X, xstride, ldk, Yh, Yf, ystride, ldy, ymode, g, be, mu, va,
          biasO, biasBO, colsum, resid, rstride, ldr, residMode, act, O, I, NN, nb);
    }
  };

  // ---- weights -> f16 ----
  cvt(w2, w2h, 128, 128, 128);
  for (int i = 0; i < 4; ++i) {
    cvt(wqk[i], wqkh[i], 32, 128, 128);
    cvt(wv[i], wvh[i], 128, 128, 128);
    cvt(wt[i], wth[i], 128, 128, 128);
  }
  cvt(wf, wfh, 1024, 512, 512);
  cvt(ws1, ws1ah, 512, 1024, 3136);
  cvt(ws2, ws2h, 256, 512, 512);
  cvt(ws3, ws3h, 50, 256, 256);

  // ---- stem ----
  {
    int cnt = B * C * N;
    stem_kernel<<<(cnt + 255) / 256, 256, 0, stream>>>(points, w1, bn1g, bn1b, bn1m, bn1v, x1T, B, N);
  }
  gemm(w2h, 0, x1T, (long)N * C, C, x0T, nullptr, (long)N * C, C, 1,
       bn2g, bn2b, bn2m, bn2v, nullptr, nullptr, nullptr, nullptr, 0, 0, 0, 1,
       128, 128, N, B);

  // ---- 4 offset-attention blocks ----
  for (int i = 0; i < 4; ++i) {
    const h16* xin = (i == 0) ? x0T : (xcatT + (size_t)(i - 1) * 128);
    long xinStride = (i == 0) ? (long)N * 128 : (long)N * 512;
    int  xinLd     = (i == 0) ? 128 : 512;
    gemm(wqkh[i], 0, xin, xinStride, xinLd, qkT, nullptr, (long)N * 32, 32, 1,
         nullptr, nullptr, nullptr, nullptr, nullptr, nullptr, nullptr, nullptr, 0, 0, 0, 0,
         32, 128, N, B);
    gemm(wvh[i], 0, xin, xinStride, xinLd, vh, nullptr, (long)128 * N, 0, 0,
         nullptr, nullptr, nullptr, nullptr, nullptr, nullptr, nullptr, nullptr, 0, 0, 0, 0,
         128, 128, N, B);
    // energy = qk^T qk (symmetric)
    gemm(qkT, (long)N * 32, qkT, (long)N * 32, 32, Eb, nullptr, (long)N * N, N, 1,
         nullptr, nullptr, nullptr, nullptr, nullptr, nullptr, nullptr, nullptr, 0, 0, 0, 0,
         N, 32, N, B);
    rowstats_kernel<<<(B * N) / 8, 256, 0, stream>>>(Eb, rmax, rinv, N, B);
    softmax_applyT_kernel<<<(B * N) / 8, 256, 0, stream>>>(Eb, rmax, rinv, csb, N, B);
    // t = x - (v @ attn)/(1e-9+colsum)
    gemm(vh, (long)128 * N, Eb, (long)N * N, N, tT, nullptr, (long)N * 128, 128, 1,
         nullptr, nullptr, nullptr, nullptr, nullptr, nullptr, csb,
         xin, xinStride, xinLd, 2, 0, 128, N, N, B);
    // x_out = x + relu(bn(wt @ t)) -> xcatT slice i
    gemm(wth[i], 0, tT, (long)N * 128, 128, xcatT + (size_t)i * 128, nullptr, (long)N * 512, 512, 1,
         bg[i], bbt[i], bm[i], bv[i], nullptr, nullptr, nullptr,
         xin, xinStride, xinLd, 1, 1, 128, 128, N, B);
  }

  // ---- xf = lrelu(bnf(wf @ xcat)) ----
  gemm(wfh, 0, xcatT, (long)N * 512, 512, xfT, nullptr, (long)N * 1024, 1024, 1,
       bnfg, bnfb, bnfm, bnfv, nullptr, nullptr, nullptr, nullptr, 0, 0, 0, 2,
       1024, 512, N, B);

  pool_kernel<<<(B * 1024 + 255) / 256, 256, 0, stream>>>(xfT, xmax, xavg, 1024, N, B);
  label_kernel<<<(B * 64 + 255) / 256, 256, 0, stream>>>(cls, wl, bnlg, bnlb, bnlm, bnlv, lblf, B);
  bias2_kernel<<<(B * 512 + 255) / 256, 256, 0, stream>>>(ws1, bs1, xmax, xavg, lblf, bias2, B);

  // ---- head ----
  gemm(ws1ah, 0, xfT, (long)N * 1024, 1024, y1T, nullptr, (long)N * 512, 512, 1,
       bns1g, bns1b, bns1m, bns1v, nullptr, bias2, nullptr, nullptr, 0, 0, 0, 1,
       512, 1024, N, B);
  gemm(ws2h, 0, y1T, (long)N * 512, 512, y2T, nullptr, (long)N * 256, 256, 1,
       bns2g, bns2b, bns2m, bns2v, bs2, nullptr, nullptr, nullptr, 0, 0, 0, 1,
       256, 512, N, B);
  gemm(ws3h, 0, y2T, (long)N * 256, 256, nullptr, y3T, (long)N * 50, 50, 1,
       nullptr, nullptr, nullptr, nullptr, bs3, nullptr, nullptr, nullptr, 0, 0, 0, 0,
       50, 256, N, B);
  logsoftmax_kernel<<<(B * N + 255) / 256, 256, 0, stream>>>(y3T, (float*)d_out, N, B);
}
